// InformerDecoder_52922587021394
// MI455X (gfx1250) — matrix operand order repaired and checked by reading, not compiled, not run on hardware
//
#include <hip/hip_runtime.h>
#include <hip/hip_bf16.h>

// ---------------------------------------------------------------------------
// Informer forward for MI455X (gfx1250, wave32, WMMA + async-to-LDS staging).
// Dense matmuls: v_wmma_f32_16x16x32_bf16 (bf16 operands, f32 accumulate).
// Tile staging: global_load_async_to_lds_b128 (ASYNCcnt) with double buffering
// so HBM->LDS traffic overlaps the matrix pipe.
// ---------------------------------------------------------------------------

typedef __bf16 bf16_t;
typedef __attribute__((ext_vector_type(16))) __bf16 v16bf;
typedef __attribute__((ext_vector_type(8)))  float  v8f;

#define ZERO8 ((v8f){0.f,0.f,0.f,0.f,0.f,0.f,0.f,0.f})

// Low 32 bits of a generic pointer to LDS == the DS byte offset
// (ISA 10.2: the LDS aperture uses the low 32 address bits as the LDS address).
__device__ inline unsigned lds_addr32(const void* p) {
  return (unsigned)(unsigned long long)p;
}

// One wave-instruction async copy: 16B per lane, global -> LDS.
__device__ inline void async_copy_b128(const void* gptr, void* lptr) {
  unsigned l = lds_addr32(lptr);
  unsigned long long g = (unsigned long long)gptr;
  asm volatile("global_load_async_to_lds_b128 %0, %1, off"
               :: "v"(l), "v"(g) : "memory");
}
__device__ inline void wait_async0() {
  asm volatile("s_wait_asynccnt 0x0" ::: "memory");
}
__device__ inline void wait_async8() {
  asm volatile("s_wait_asynccnt 0x8" ::: "memory");
}

// ---- WMMA fragment builders (contiguous k = 16*half + e in A and B, wave32) ---------
// A 16x32 bf16 from a row-major f32 tile: element e=2r+j ->
//   K = 16*half + 2*r + j,  M = lane&15.
__device__ inline v16bf load_a_frag_f32(const float* base, int ld, int lane) {
  int half = lane >> 4, l16 = lane & 15;
  const float* row = base + (size_t)l16 * ld;
  v16bf f;
#pragma unroll
  for (int r = 0; r < 8; ++r) {
    int k = 16 * half + 2 * r;
    f[2 * r]     = (bf16_t)row[k];
    f[2 * r + 1] = (bf16_t)row[k + 1];
  }
  return f;
}
// A 16x32 from a bf16 tile (used for the softmax P tile).
__device__ inline v16bf load_a_frag_bf16(const bf16_t* base, int ld, int lane) {
  int half = lane >> 4, l16 = lane & 15;
  const bf16_t* row = base + (size_t)l16 * ld;
  v16bf f;
#pragma unroll
  for (int r = 0; r < 8; ++r) {
    int k = 16 * half + 2 * r;
    f[2 * r]     = row[k];
    f[2 * r + 1] = row[k + 1];
  }
  return f;
}
// B 32x16 from a row-major KxN f32 tile: frag[e] = B[16*half+e][lane&15].
__device__ inline v16bf load_b_frag_f32(const float* base, int ld, int lane) {
  int half = lane >> 4, l16 = lane & 15;
  v16bf f;
#pragma unroll
  for (int e = 0; e < 16; ++e)
    f[e] = (bf16_t)base[(size_t)(16 * half + e) * ld + l16];
  return f;
}
// B 32x16 built as tile^T from a row-major NxK f32 tile:
// frag[e] = tile[lane&15][16*half + e].
__device__ inline v16bf load_bT_frag_f32(const float* base, int ld, int lane) {
  int half = lane >> 4, l16 = lane & 15;
  const float* row = base + (size_t)l16 * ld;
  v16bf f;
#pragma unroll
  for (int e = 0; e < 16; ++e) f[e] = (bf16_t)row[16 * half + e];
  return f;
}

__device__ inline v8f wmma_bf16(v16bf a, v16bf b, v8f c) {
  return __builtin_amdgcn_wmma_f32_16x16x32_bf16(false, a, false, b,
                                                 (short)0, c, false, false);
}

__device__ inline float gelu_f(float x) {
  return 0.5f * x * (1.f + erff(x * 0.70710678118654752f));
}

// ---------------------------------------------------------------------------
// GEMM: C[M,N] = act( A[M,K] @ W[N,K]^T + bias[N] ).
// 128 threads = 4 waves; block computes a 64x64 C tile (one 32x32 per wave).
// A/W tiles double-buffered in LDS, filled by async b128 copies.
// ---------------------------------------------------------------------------
__global__ __launch_bounds__(128) void gemm_linear_wmma(
    const float* __restrict__ A, const float* __restrict__ W,
    const float* __restrict__ bias, float* __restrict__ C,
    int M, int N, int K, int act) {
  __shared__ float Af[2][64][36];  // 36-f32 stride: 16B-aligned rows, bank-spread
  __shared__ float Wf[2][64][36];

  const int tid = threadIdx.x;
  const int wave = tid >> 5, lane = tid & 31;
  const int half = lane >> 4, l16 = lane & 15;
  const int m0 = blockIdx.x * 64, n0 = blockIdx.y * 64;
  const int wm = (wave & 1) * 32, wn = (wave >> 1) * 32;

  // Issue one 64x32 f32 tile of A and of W into LDS buffer `buf`.
  // 512 chunks of 16B per array; 4 per thread => 8 async instrs per wave.
  auto issue = [&](int buf, int k0) {
    for (int i = tid; i < 512; i += 128) {
      int r = i >> 3, c4 = (i & 7) * 4;
      async_copy_b128(A + (size_t)(m0 + r) * K + k0 + c4, &Af[buf][r][c4]);
      async_copy_b128(W + (size_t)(n0 + r) * K + k0 + c4, &Wf[buf][r][c4]);
    }
  };

  v8f acc[2][2];
  acc[0][0] = ZERO8; acc[0][1] = ZERO8; acc[1][0] = ZERO8; acc[1][1] = ZERO8;

  issue(0, 0);
  for (int k0 = 0; k0 < K; k0 += 32) {
    int cur = (k0 >> 5) & 1;
    if (k0 + 32 < K) { issue(cur ^ 1, k0 + 32); wait_async8(); }
    else             { wait_async0(); }
    __syncthreads();

    v16bf b0 = load_bT_frag_f32(&Wf[cur][wn + 0][0],  36, lane);
    v16bf b1 = load_bT_frag_f32(&Wf[cur][wn + 16][0], 36, lane);
#pragma unroll
    for (int ms = 0; ms < 2; ++ms) {
      v16bf af = load_a_frag_f32(&Af[cur][wm + 16 * ms][0], 36, lane);
      acc[ms][0] = wmma_bf16(af, b0, acc[ms][0]);
      acc[ms][1] = wmma_bf16(af, b1, acc[ms][1]);
    }
    __syncthreads();  // everyone done reading `cur` before it is re-filled
  }

#pragma unroll
  for (int ms = 0; ms < 2; ++ms)
#pragma unroll
    for (int ns = 0; ns < 2; ++ns) {
      int n = n0 + wn + ns * 16 + l16;
      float bv = bias ? bias[n] : 0.f;
#pragma unroll
      for (int r = 0; r < 8; ++r) {
        int m = m0 + wm + ms * 16 + r + 8 * half;
        float v = acc[ms][ns][r] + bv;
        if (act == 1) v = gelu_f(v);
        C[(size_t)m * N + n] = v;
      }
    }
}

// ---------------------------------------------------------------------------
// Fused flash attention: one wave handles 16 queries for one (b,h).
// Q,K,V,O layout [B, L, H, 64]; softmax(QK^T/8)V, optional causal mask.
// K/V chunk staging uses async b128 copies.
// ---------------------------------------------------------------------------
__global__ __launch_bounds__(32) void attn_wmma(
    const float* __restrict__ Q, const float* __restrict__ K,
    const float* __restrict__ V, float* __restrict__ O,
    int Bn, int H, int Lq, int S, int causal) {
  const int DH = 64;
  __shared__ float  Qsf[16][68];
  __shared__ float  Ksf[32][68];
  __shared__ float  Vsf[32][68];
  __shared__ bf16_t Ps[16][34];

  const int lane = threadIdx.x;
  const int half = lane >> 4, l16 = lane & 15;
  const int ntile = Lq / 16;
  const int qt = blockIdx.x % ntile;
  const int bh = blockIdx.x / ntile;
  const int h = bh % H, b = bh / H;
  const int q0 = qt * 16;
  const float scale = 0.125f;  // 1/sqrt(64)

  // stage Q (16x64 f32): 256 16B chunks, 8 per lane
  for (int i = lane; i < 256; i += 32) {
    int r = i >> 4, c4 = (i & 15) * 4;
    async_copy_b128(Q + (((size_t)b * Lq + q0 + r) * H + h) * DH + c4,
                    &Qsf[r][c4]);
  }
  wait_async0();
  __syncthreads();

  v16bf qf[2];
  qf[0] = load_a_frag_f32(&Qsf[0][0],  68, lane);
  qf[1] = load_a_frag_f32(&Qsf[0][32], 68, lane);

  float mrow[8], lrow[8];
  v8f oacc[4];
#pragma unroll
  for (int r = 0; r < 8; ++r) { mrow[r] = -1e30f; lrow[r] = 0.f; }
#pragma unroll
  for (int t = 0; t < 4; ++t) oacc[t] = ZERO8;

  for (int s0 = 0; s0 < S; s0 += 32) {
    if (causal && s0 > q0 + 15) break;
    for (int i = lane; i < 512; i += 32) {
      int r = i >> 4, c4 = (i & 15) * 4;
      size_t g = (((size_t)b * S + s0 + r) * H + h) * DH + c4;
      async_copy_b128(K + g, &Ksf[r][c4]);
      async_copy_b128(V + g, &Vsf[r][c4]);
    }
    wait_async0();
    __syncthreads();

    // S tile = Q (16x64) * K^T (64x32) -> two 16x16 accumulators
    v8f st[2]; st[0] = ZERO8; st[1] = ZERO8;
#pragma unroll
    for (int c = 0; c < 2; ++c)
#pragma unroll
      for (int t = 0; t < 2; ++t)
        st[t] = wmma_bf16(qf[c],
                          load_bT_frag_f32(&Ksf[16 * t][32 * c], 68, lane),
                          st[t]);

    // online softmax over this 32-wide chunk
#pragma unroll
    for (int r = 0; r < 8; ++r) {
      int row = q0 + r + 8 * half;
      float x0 = st[0][r] * scale, x1 = st[1][r] * scale;
      if (causal) {
        if (s0 + l16 > row)      x0 = -1e30f;
        if (s0 + 16 + l16 > row) x1 = -1e30f;
      }
      float mx = fmaxf(x0, x1);
#pragma unroll
      for (int off = 1; off < 16; off <<= 1) mx = fmaxf(mx, __shfl_xor(mx, off, 16));
      float mnew = fmaxf(mrow[r], mx);
      float p0 = __expf(x0 - mnew), p1 = __expf(x1 - mnew);
      float ls = p0 + p1;
#pragma unroll
      for (int off = 1; off < 16; off <<= 1) ls += __shfl_xor(ls, off, 16);
      float corr = __expf(mrow[r] - mnew);
      lrow[r] = lrow[r] * corr + ls;
      mrow[r] = mnew;
#pragma unroll
      for (int t = 0; t < 4; ++t) oacc[t][r] *= corr;
      Ps[r + 8 * half][l16]      = (bf16_t)p0;
      Ps[r + 8 * half][16 + l16] = (bf16_t)p1;
    }
    __syncthreads();

    // O += P (16x32) * V (32x64)
    v16bf pa = load_a_frag_bf16(&Ps[0][0], 34, lane);
#pragma unroll
    for (int t = 0; t < 4; ++t)
      oacc[t] = wmma_bf16(pa, load_b_frag_f32(&Vsf[0][16 * t], 68, lane),
                          oacc[t]);
    __syncthreads();
  }

#pragma unroll
  for (int t = 0; t < 4; ++t)
#pragma unroll
    for (int r = 0; r < 8; ++r) {
      int m = q0 + r + 8 * half;
      int n = 16 * t + l16;
      O[(((size_t)b * Lq + m) * H + h) * DH + n] =
          oacc[t][r] / fmaxf(lrow[r], 1e-20f);
    }
}

// ---------------------------------------------------------------------------
// Data embedding: circular conv1d(x, Wemb) + positional + time encodings.
// ---------------------------------------------------------------------------
__global__ __launch_bounds__(128) void embed_kernel(
    const float* __restrict__ x, const int* __restrict__ mark,
    const float* __restrict__ Wc, float* __restrict__ out,
    int Bn, int L, int Cin, int D) {
  int bt = blockIdx.x;
  int b = bt / L, t = bt % L;
  for (int d = threadIdx.x; d < D; d += blockDim.x) {
    float acc = 0.f;
    for (int w = 0; w < 3; ++w) {
      int ts = t - 1 + w; ts = (ts + L) % L;
      const float* xr = x + ((size_t)b * L + ts) * Cin;
      const float* wr = Wc + ((size_t)w * Cin) * D + d;
      for (int c = 0; c < Cin; ++c) acc += xr[c] * wr[(size_t)c * D];
    }
    int pair = d >> 1;
    float freq = __expf(-logf(10000.f) * (2.f * pair) / (float)D);
    float pe = (d & 1) ? __cosf(t * freq) : __sinf(t * freq);
    float tm = 0.f;
    for (int i = 0; i < 4; ++i) {
      int pos = mark[((size_t)b * L + t) * 4 + i];
      tm += (d & 1) ? __cosf(pos * freq) : __sinf(pos * freq);
    }
    out[((size_t)b * L + t) * D + d] = acc + pe + tm;
  }
}

// ---------------------------------------------------------------------------
__device__ inline float block_reduce_sum(float v, float* red) {
  __syncthreads();
  int lane = threadIdx.x & 31, wid = threadIdx.x >> 5;
#pragma unroll
  for (int off = 16; off > 0; off >>= 1) v += __shfl_down(v, off, 32);
  if (lane == 0) red[wid] = v;
  __syncthreads();
  int nw = (blockDim.x + 31) >> 5;
  float r = (threadIdx.x < nw) ? red[threadIdx.x] : 0.f;
  if (wid == 0) {
#pragma unroll
    for (int off = 16; off > 0; off >>= 1) r += __shfl_down(r, off, 32);
    if (lane == 0) red[0] = r;
  }
  __syncthreads();
  return red[0];
}

// LayerNorm(x + h) * g + b ; h may be null; out may alias x.
__global__ __launch_bounds__(256) void ln_kernel(
    const float* __restrict__ x, const float* __restrict__ h,
    const float* __restrict__ g, const float* __restrict__ bta,
    float* __restrict__ out, int D) {
  __shared__ float red[32];
  size_t row = blockIdx.x;
  const float* xr = x + row * D;
  const float* hr = h ? h + row * D : nullptr;
  float s = 0.f;
  for (int d = threadIdx.x; d < D; d += blockDim.x)
    s += xr[d] + (hr ? hr[d] : 0.f);
  float mean = block_reduce_sum(s, red) / (float)D;
  float s2 = 0.f;
  for (int d = threadIdx.x; d < D; d += blockDim.x) {
    float v = xr[d] + (hr ? hr[d] : 0.f) - mean;
    s2 += v * v;
  }
  float var = block_reduce_sum(s2, red) / (float)D;
  float inv = rsqrtf(var + 1e-5f);
  for (int d = threadIdx.x; d < D; d += blockDim.x) {
    float v = xr[d] + (hr ? hr[d] : 0.f);
    out[row * D + d] = (v - mean) * inv * g[d] + bta[d];
  }
}

// circular conv over length with [3, D, D] weights + bias
__global__ __launch_bounds__(128) void circconv_kernel(
    const float* __restrict__ x, const float* __restrict__ W,
    const float* __restrict__ bias, float* __restrict__ y, int L, int D) {
  int bt = blockIdx.x;
  int b = bt / L, t = bt % L;
  for (int d = threadIdx.x; d < D; d += blockDim.x) {
    float acc = bias[d];
    for (int w = 0; w < 3; ++w) {
      int ts = t - 1 + w; ts = (ts + L) % L;
      const float* xr = x + ((size_t)b * L + ts) * D;
      const float* wr = W + ((size_t)w * D) * D + d;
      for (int c = 0; c < D; ++c) acc += xr[c] * wr[(size_t)c * D];
    }
    y[((size_t)b * L + t) * D + d] = acc;
  }
}

__global__ __launch_bounds__(256) void bnstats_kernel(
    const float* __restrict__ y, float* __restrict__ stats, int rows, int D) {
  __shared__ float red[32];
  int c = blockIdx.x;
  float s = 0.f, s2 = 0.f;
  for (int r = threadIdx.x; r < rows; r += blockDim.x) {
    float v = y[(size_t)r * D + c];
    s += v; s2 += v * v;
  }
  float sum = block_reduce_sum(s, red);
  float sq  = block_reduce_sum(s2, red);
  if (threadIdx.x == 0) {
    float mu = sum / (float)rows;
    stats[c] = mu;
    stats[D + c] = sq / (float)rows - mu * mu;
  }
}

// batchnorm + ELU + maxpool(window 3, stride 2, pad 1): out length L/2
__global__ __launch_bounds__(256) void bn_elu_pool_kernel(
    const float* __restrict__ y, const float* __restrict__ stats,
    const float* __restrict__ g, const float* __restrict__ bta,
    float* __restrict__ out, int L, int D) {
  int Lo = L / 2;
  size_t total = (size_t)4 * Lo * D;
  for (size_t idx = (size_t)blockIdx.x * blockDim.x + threadIdx.x; idx < total;
       idx += (size_t)gridDim.x * blockDim.x) {
    int c = idx % D;
    size_t rest = idx / D;
    int to = rest % Lo;
    int b = rest / Lo;
    float mu = stats[c], var = stats[D + c];
    float inv = rsqrtf(var + 1e-5f);
    float best = -1e30f;
    for (int w = -1; w <= 1; ++w) {
      int ti = 2 * to + w;
      if (ti < 0 || ti >= L) continue;
      float v = (y[(((size_t)b * L + ti)) * D + c] - mu) * inv * g[c] + bta[c];
      v = (v > 0.f) ? v : (__expf(v) - 1.f);  // ELU
      best = fmaxf(best, v);
    }
    out[(((size_t)b * Lo + to)) * D + c] = best;
  }
}

// final: proj(dec) -> concat with x_enc[:,:,:96] -> skip linear [24]
__global__ __launch_bounds__(128) void proj_skip_kernel(
    const float* __restrict__ dec, const float* __restrict__ x_enc,
    const float* __restrict__ projW, const float* __restrict__ projB,
    const float* __restrict__ skipW, const float* __restrict__ skipB,
    float* __restrict__ out, int D, int Cin, int P) {
  __shared__ float red[32];
  __shared__ float s_in[97];
  size_t row = blockIdx.x;
  float s = 0.f;
  for (int d = threadIdx.x; d < D; d += blockDim.x)
    s += dec[row * D + d] * projW[d];
  float pv = block_reduce_sum(s, red) + projB[0];
  if (threadIdx.x == 0) s_in[0] = pv;
  for (int c = threadIdx.x; c < Cin; c += blockDim.x)
    s_in[1 + c] = x_enc[row * Cin + c];
  __syncthreads();
  for (int p = threadIdx.x; p < P; p += blockDim.x) {
    float acc = skipB[p];
    for (int j = 0; j < Cin + 1; ++j) acc += s_in[j] * skipW[p * (Cin + 1) + j];
    out[row * P + p] = acc;
  }
}

// ---------------------------------------------------------------------------
// Host orchestration.  Flattened input index map (setup_inputs() insertion
// order): 0 x_enc, 1 enc_emb, 2 dec_emb; enc layer i base=3+16i
// (qW qb kW kb vW vb oW ob ff1W ff1b ff2W ff2b ln1g ln1b ln2g ln2b);
// enc_convs j base=51+4j (W b bn_g bn_b); enc_norm 59/60;
// dec layer j base=61+26j (self 0..7, cross 8..15, ff1 16/17, ff2 18/19,
// ln1 20/21, ln2 22/23, ln3 24/25); dec_norm 113/114; proj 115/116;
// skip 117/118; 119 x_mark_enc; 120 x_mark_dec.
// ---------------------------------------------------------------------------
extern "C" void kernel_launch(void* const* d_in, const int* in_sizes, int n_in,
                              void* d_out, int out_size, void* d_ws, size_t ws_size,
                              hipStream_t stream) {
  (void)in_sizes; (void)n_in; (void)out_size; (void)ws_size;
  auto F = [&](int i) { return (const float*)d_in[i]; };
  const float* x_enc = F(0);
  const int* mark = (const int*)d_in[119];

  const int Bn = 4, L0 = 1024, D = 512, H = 8, CIN = 96, PRED = 24;
  float* ws = (float*)d_ws;
  const size_t BLD = (size_t)Bn * L0 * D;
  float* X   = ws + 0 * BLD;
  float* Qb  = ws + 1 * BLD;
  float* Kb  = ws + 2 * BLD;
  float* Vb  = ws + 3 * BLD;
  float* Ob  = ws + 4 * BLD;
  float* T1  = ws + 5 * BLD;
  float* T2  = ws + 6 * BLD;
  float* DEC = ws + 7 * BLD;
  float* ENC = ws + 8 * BLD;
  float* CY  = ws + 9 * BLD;
  float* ST  = ws + 10 * BLD;  // bn stats (2*D)

  auto linear = [&](const float* A, int wi, int bi, float* C, int M, int act) {
    gemm_linear_wmma<<<dim3(M / 64, D / 64), dim3(128), 0, stream>>>(
        A, F(wi), F(bi), C, M, D, D, act);
  };
  auto lnorm = [&](const float* x, const float* h, int gi, int bi, float* o, int rows) {
    ln_kernel<<<dim3(rows), dim3(256), 0, stream>>>(x, h, F(gi), F(bi), o, D);
  };
  auto attn = [&](const float* Q, const float* K, const float* V, float* O,
                  int Lq, int S, int causal) {
    attn_wmma<<<dim3(Bn * H * (Lq / 16)), dim3(32), 0, stream>>>(
        Q, K, V, O, Bn, H, Lq, S, causal);
  };

  // ---- Encoder ----
  embed_kernel<<<dim3(Bn * L0), dim3(128), 0, stream>>>(x_enc, mark, F(1), X,
                                                        Bn, L0, CIN, D);
  int Le = L0;
  for (int i = 0; i < 3; ++i) {
    int base = 3 + 16 * i;
    int M = Bn * Le;
    linear(X, base + 0, base + 1, Qb, M, 0);
    linear(X, base + 2, base + 3, Kb, M, 0);
    linear(X, base + 4, base + 5, Vb, M, 0);
    attn(Qb, Kb, Vb, Ob, Le, Le, 0);
    linear(Ob, base + 6, base + 7, T1, M, 0);
    lnorm(X, T1, base + 12, base + 13, X, M);
    linear(X, base + 8, base + 9, T1, M, /*gelu*/ 1);
    linear(T1, base + 10, base + 11, T2, M, 0);
    lnorm(X, T2, base + 14, base + 15, X, M);
    if (i < 2) {
      int cb = 51 + 4 * i;
      circconv_kernel<<<dim3(Bn * Le), dim3(128), 0, stream>>>(X, F(cb), F(cb + 1),
                                                               CY, Le, D);
      bnstats_kernel<<<dim3(D), dim3(256), 0, stream>>>(CY, ST, Bn * Le, D);
      bn_elu_pool_kernel<<<dim3(2048), dim3(256), 0, stream>>>(
          CY, ST, F(cb + 2), F(cb + 3), X, Le, D);
      Le /= 2;
    }
  }
  lnorm(X, nullptr, 59, 60, ENC, Bn * Le);  // Le == 256

  // ---- Decoder ----
  embed_kernel<<<dim3(Bn * L0), dim3(128), 0, stream>>>(x_enc, mark, F(2), DEC,
                                                        Bn, L0, CIN, D);
  const int Md = Bn * L0;
  for (int j = 0; j < 2; ++j) {
    int base = 61 + 26 * j;
    // self-attention (causal)
    linear(DEC, base + 0, base + 1, Qb, Md, 0);
    linear(DEC, base + 2, base + 3, Kb, Md, 0);
    linear(DEC, base + 4, base + 5, Vb, Md, 0);
    attn(Qb, Kb, Vb, Ob, L0, L0, 1);
    linear(Ob, base + 6, base + 7, T1, Md, 0);
    lnorm(DEC, T1, base + 20, base + 21, DEC, Md);
    // cross-attention vs encoder output (S = 256)
    linear(DEC, base + 8, base + 9, Qb, Md, 0);
    linear(ENC, base + 10, base + 11, Kb, Bn * Le, 0);
    linear(ENC, base + 12, base + 13, Vb, Bn * Le, 0);
    attn(Qb, Kb, Vb, Ob, L0, Le, 0);
    linear(Ob, base + 14, base + 15, T1, Md, 0);
    lnorm(DEC, T1, base + 22, base + 23, DEC, Md);
    // FFN
    linear(DEC, base + 16, base + 17, T1, Md, /*gelu*/ 1);
    linear(T1, base + 18, base + 19, T2, Md, 0);
    lnorm(DEC, T2, base + 24, base + 25, DEC, Md);
  }
  lnorm(DEC, nullptr, 113, 114, T1, Md);

  // ---- proj + skip ----
  proj_skip_kernel<<<dim3(Bn * L0), dim3(128), 0, stream>>>(
      T1, x_enc, F(115), F(116), F(117), F(118), (float*)d_out, D, CIN, PRED);
}
